// MyLSTM_26706106646797
// MI455X (gfx1250) — matrix operand order repaired and checked by reading, not compiled, not run on hardware
//
#include <hip/hip_runtime.h>

typedef __attribute__((ext_vector_type(16))) _Float16 v16h;
typedef __attribute__((ext_vector_type(8)))  _Float16 v8h;
typedef __attribute__((ext_vector_type(4)))  _Float16 v4h;
typedef __attribute__((ext_vector_type(8)))  float    v8f;
typedef __attribute__((ext_vector_type(2)))  int      v2i;

#define HID   64
#define GATES 256          // 4*HID
#define KDIM  128          // [x(64, padded) | h(64)]
#define TSTEPS 2048
#define BATCH 256
#define NBT   16           // batch tiles of 16 rows
#define NLAY  5
#define RING  64           // ring depth in steps (power of 2)
#define RINGH (BATCH * RING * HID)   // f16 elements per inter-layer ring

// Native CDNA5 v_tanh_f32 if the toolchain exposes it; safe fallback otherwise.
#if defined(__has_builtin)
#if __has_builtin(__builtin_amdgcn_tanhf)
#define FAST_TANH(x) __builtin_amdgcn_tanhf(x)
#endif
#if __has_builtin(__builtin_amdgcn_global_load_async_to_lds_b64) && \
    __has_builtin(__builtin_amdgcn_s_wait_asynccnt)
#define HAVE_ASYNC_LDS 1
#endif
#endif
#ifndef FAST_TANH
#define FAST_TANH(x) tanhf(x)
#endif

__device__ __forceinline__ float fast_tanh(float x) { return FAST_TANH(x); }
// sigmoid(x) = 0.5*tanh(x/2) + 0.5  -> one TRANS + one FMA
__device__ __forceinline__ float sigm(float x) {
    return fmaf(0.5f, FAST_TANH(0.5f * x), 0.5f);
}

// Persistent pipelined LSTM: grid (16 batch-tiles, 5 layers), block 256 (8 waves).
// Layer l consumes ring[l-1] (or x for l==0) and produces ring[l] (or lastH for l==4).
// Rings hold h in f16 (WMMA input precision) to halve traffic and skip conversions.
__global__ void __launch_bounds__(256)
lstm_pipeline_kernel(const float* __restrict__ x,
                     const float* __restrict__ W_ih0,      // [256,45]
                     const float* __restrict__ W_ih_rest,  // [4,256,64]
                     const float* __restrict__ W_hh,       // [5,256,64]
                     const float* __restrict__ b_ih,       // [5,256]
                     const float* __restrict__ b_hh,       // [5,256]
                     _Float16* __restrict__ rings,         // 4 * RINGH halves
                     int*      __restrict__ flags,         // [0..79] prod, [80..159] cons
                     float*    __restrict__ lastH)         // [256,64]
{
    __shared__ __attribute__((aligned(16))) _Float16 Abuf[16][KDIM]; // [row][k]
    __shared__ float gbuf[16][GATES];                                // raw gates

    const int bt = blockIdx.x;        // batch tile 0..15
    const int l  = blockIdx.y;        // layer 0..4
    const int b0 = bt * 16;
    const int tid  = threadIdx.x;
    const int lane = tid & 31;
    const int wave = tid >> 5;

    const int D  = (l == 0) ? 45 : 64;           // valid input width
    const float* Wih = (l == 0) ? W_ih0 : (W_ih_rest + (size_t)(l - 1) * GATES * HID);
    const float* Whh = W_hh + (size_t)l * GATES * HID;
    const _Float16* ringIn  = (l > 0) ? (rings + (size_t)(l - 1) * RINGH) : nullptr;
    _Float16*       ringOut = (l < 4) ? (rings + (size_t)l * RINGH)       : nullptr;

    int* prodPrev = (l > 0) ? &flags[(l - 1) * NBT + bt] : nullptr;
    int* prodSelf = (l < 4) ? &flags[l * NBT + bt]       : nullptr;
    int* consSelf = (l > 0) ? &flags[80 + l * NBT + bt]  : nullptr;
    int* consNext = (l < 4) ? &flags[80 + (l + 1) * NBT + bt] : nullptr;

    // ---- zero A-buffer (x-pad stays 0; h starts at 0) ----
    for (int i = tid; i < 16 * KDIM; i += 256)
        ((_Float16*)Abuf)[i] = (_Float16)0.0f;

    // ---- preload B fragments (weights) + bias into registers, reused all T steps ----
    // B layout (f16 32x16 per k-chunk): lane holds column n = tile*16 + (lane&15),
    // k = kc*32 + (lane>>4)*16 + e, e = 0..15 (contiguous in memory row).
    const int ncol  = lane & 15;
    const int khalf = (lane >> 4) << 4;
    v16h Bfrag[2][4];
    float bias2[2];
#pragma unroll
    for (int ti = 0; ti < 2; ++ti) {
        const int tile = wave * 2 + ti;
        const int r = tile * 16 + ncol;     // gate row 0..255
        bias2[ti] = b_ih[l * GATES + r] + b_hh[l * GATES + r];
#pragma unroll
        for (int kc = 0; kc < 4; ++kc) {
            v16h f;
#pragma unroll
            for (int e = 0; e < 16; ++e) {
                const int kk = kc * 32 + khalf + e;
                float w;
                if (kk < 64) w = (kk < D) ? Wih[(size_t)r * D + kk] : 0.0f;
                else         w = Whh[(size_t)r * HID + (kk - 64)];
                f[e] = (_Float16)w;
            }
            Bfrag[ti][kc] = f;
        }
    }
    __syncthreads();

    // pointwise-phase mapping: each thread owns (row, 4 hidden cols)
    const int prow = tid >> 4;           // 0..15
    const int pcb  = (tid & 15) * 4;     // 0..60
    float cst[4] = {0.f, 0.f, 0.f, 0.f};

    // A-fragment mapping
    const int am = lane & 15;
    const int ab = (lane >> 4) * 16;
    const int mb = (lane >= 16) ? 8 : 0;

    for (int t = 0; t < TSTEPS; ++t) {
        const int tmod = t & (RING - 1);

        // ---- flow control (thread 0 spins; agent-scope acquire) ----
        if (tid == 0) {
            if (l > 0) {
                while (__hip_atomic_load(prodPrev, __ATOMIC_ACQUIRE,
                                         __HIP_MEMORY_SCOPE_AGENT) < t + 1)
                    __builtin_amdgcn_s_sleep(1);
            }
            if (l < 4 && t >= RING) {
                while (__hip_atomic_load(consNext, __ATOMIC_ACQUIRE,
                                         __HIP_MEMORY_SCOPE_AGENT) < t - RING + 1)
                    __builtin_amdgcn_s_sleep(1);
            }
        }
        __syncthreads();
        if (l > 0)  // make producer's ring data visible to all waves/CUs of this WGP
            __builtin_amdgcn_fence(__ATOMIC_ACQUIRE, "agent");

        // ---- phase 1: stage x_t into LDS (f16) ----
        if (l == 0) {
            const float* xin = x + ((size_t)(b0 + prow) * TSTEPS + t) * 45;
#pragma unroll
            for (int j = 0; j < 4; ++j) {
                const int col = pcb + j;
                if (col < 45) Abuf[prow][col] = (_Float16)xin[col];
            }
        } else {
            const _Float16* src = ringIn +
                (((size_t)(b0 + prow) * RING + tmod) * HID + pcb);
#ifdef HAVE_ASYNC_LDS
            // async DMA: 8 bytes (4 halves) straight from L2-resident ring into LDS
            __builtin_amdgcn_global_load_async_to_lds_b64(
                (__attribute__((address_space(1))) v2i*)src,
                (__attribute__((address_space(3))) v2i*)&Abuf[prow][pcb],
                0, 0);
            __builtin_amdgcn_s_wait_asynccnt(0);
#else
            const v4h v = *(const v4h*)src;
            *(v4h*)&Abuf[prow][pcb] = v;
#endif
        }
        __syncthreads();
        if (l > 0 && tid == 0)
            __hip_atomic_store(consSelf, t + 1, __ATOMIC_RELAXED, __HIP_MEMORY_SCOPE_AGENT);

        // ---- phase 2: g = [x|h] @ Wcat^T + b via WMMA f32<-f16 ----
        v16h a[4];
#pragma unroll
        for (int kc = 0; kc < 4; ++kc) {
            const v8h lo = *(const v8h*)&Abuf[am][kc * 32 + ab];
            const v8h hi = *(const v8h*)&Abuf[am][kc * 32 + ab + 8];
            v16h av;
#pragma unroll
            for (int e = 0; e < 8; ++e) { av[e] = lo[e]; av[e + 8] = hi[e]; }
            a[kc] = av;
        }
#pragma unroll
        for (int ti = 0; ti < 2; ++ti) {
            v8f c;
#pragma unroll
            for (int e = 0; e < 8; ++e) c[e] = bias2[ti];
#pragma unroll
            for (int kc = 0; kc < 4; ++kc)
                c = __builtin_amdgcn_wmma_f32_16x16x32_f16(
                        false, a[kc], false, Bfrag[ti][kc], (short)0, c, false, false);
            const int col = (wave * 2 + ti) * 16 + ncol;
#pragma unroll
            for (int r2 = 0; r2 < 8; ++r2) gbuf[mb + r2][col] = c[r2];
        }
        __syncthreads();

        // ---- phase 3: gate nonlinearities, state update, publish h ----
        v4h hv16;
        float4 hv;
#pragma unroll
        for (int j = 0; j < 4; ++j) {
            const int col = pcb + j;
            const float iv = gbuf[prow][col];
            const float fv = gbuf[prow][64 + col];
            const float gv = gbuf[prow][128 + col];
            const float ov = gbuf[prow][192 + col];
            const float cn = sigm(fv) * cst[j] + sigm(iv) * fast_tanh(gv);
            cst[j] = cn;
            const float h = sigm(ov) * fast_tanh(cn);
            const _Float16 h16 = (_Float16)h;
            Abuf[prow][64 + col] = h16;                // h half of next step's A
            hv16[j] = h16;
            (&hv.x)[j] = h;
        }
        if (l < 4) {
            *(v4h*)(ringOut + (((size_t)(b0 + prow) * RING + tmod) * HID + pcb)) = hv16;
            __threadfence();   // release our ring stores before flag publish
        } else if (t == TSTEPS - 1) {
            *(float4*)(lastH + (size_t)(b0 + prow) * HID + pcb) = hv;
        }
        __syncthreads();
        if (l < 4 && tid == 0)
            __hip_atomic_store(prodSelf, t + 1, __ATOMIC_RELEASE, __HIP_MEMORY_SCOPE_AGENT);
    }
}

// Head: features = gelu(lastH @ Wl^T + bl); out = relu(features @ Wo^T + bo)
// d_out layout: out [256*4] then features [256*128]
__global__ void __launch_bounds__(128)
head_kernel(const float* __restrict__ lastH,
            const float* __restrict__ Wl, const float* __restrict__ bl,
            const float* __restrict__ Wo, const float* __restrict__ bo,
            float* __restrict__ out)
{
    __shared__ float feat[128];
    const int b = blockIdx.x;
    const int f = threadIdx.x;
    float acc = bl[f];
    const float* hr = lastH + (size_t)b * HID;
    const float* wr = Wl + (size_t)f * HID;
#pragma unroll 8
    for (int k = 0; k < HID; ++k) acc += hr[k] * wr[k];
    const float ge = 0.5f * acc * (1.0f + erff(acc * 0.70710678118654752f));
    feat[f] = ge;
    out[BATCH * 4 + (size_t)b * 128 + f] = ge;
    __syncthreads();
    if (f < 4) {
        float a2 = bo[f];
        const float* wo = Wo + (size_t)f * 128;
#pragma unroll 8
        for (int k = 0; k < 128; ++k) a2 += feat[k] * wo[k];
        out[(size_t)b * 4 + f] = fmaxf(a2, 0.0f);
    }
}

extern "C" void kernel_launch(void* const* d_in, const int* in_sizes, int n_in,
                              void* d_out, int out_size, void* d_ws, size_t ws_size,
                              hipStream_t stream) {
    const float* x         = (const float*)d_in[0];
    const float* W_ih0     = (const float*)d_in[1];
    const float* W_ih_rest = (const float*)d_in[2];
    const float* W_hh      = (const float*)d_in[3];
    const float* b_ih      = (const float*)d_in[4];
    const float* b_hh      = (const float*)d_in[5];
    const float* Wl        = (const float*)d_in[6];
    const float* bl        = (const float*)d_in[7];
    const float* Wo        = (const float*)d_in[8];
    const float* bo        = (const float*)d_in[9];
    float* out = (float*)d_out;

    char* ws = (char*)d_ws;
    int*      flags = (int*)ws;                               // 160 ints (pad to 1 KB)
    float*    lastH = (float*)(ws + 1024);                    // 256*64 f32 = 64 KB
    _Float16* rings = (_Float16*)(ws + 1024 + BATCH * HID * 4); // 4 * RINGH f16 = 8 MB

    (void)hipMemsetAsync(flags, 0, 256 * sizeof(int), stream);

    dim3 grid(NBT, NLAY);
    lstm_pipeline_kernel<<<grid, dim3(256), 0, stream>>>(
        x, W_ih0, W_ih_rest, W_hh, b_ih, b_hh, rings, flags, lastH);

    head_kernel<<<dim3(BATCH), dim3(128), 0, stream>>>(lastH, Wl, bl, Wo, bo, out);
}
